// XCA_label_1_8916352106922
// MI455X (gfx1250) — compile-verified, matrix operand order checked by reading, not run on hardware
//
#include <hip/hip_runtime.h>
#include <hip/hip_bf16.h>
#include <math.h>

// ---------------------------------------------------------------------------
// Types for CDNA5 WMMA (wave32): D(16x16 f32) = A(16x32 f16) * B(32x16 f16) + C
// ---------------------------------------------------------------------------
typedef __attribute__((ext_vector_type(16))) _Float16 v16h;
typedef __attribute__((ext_vector_type(8)))  _Float16 v8h;
typedef __attribute__((ext_vector_type(8)))  float    v8f;

static constexpr int BB = 8;     // batch
static constexpr int NN = 1024;  // tokens
static constexpr int CC = 768;   // channels
static constexpr int HH = 6;     // heads
static constexpr int DD = 128;   // head dim

// Async global->LDS (CDNA5 GLOBAL_LOAD_ASYNC_TO_LDS_B128, ASYNCcnt tracked).
// Gated: fall back to synchronous copies if the toolchain lacks the builtins.
#if defined(__has_builtin)
#if __has_builtin(__builtin_amdgcn_global_load_async_to_lds_b128) && \
    __has_builtin(__builtin_amdgcn_s_wait_asynccnt)
#define HAVE_ASYNC_LDS 1
#endif
#endif
#ifndef HAVE_ASYNC_LDS
#define HAVE_ASYNC_LDS 0
#endif

#if HAVE_ASYNC_LDS
// Parameter type per hipcc diagnostic: 128-bit int vector, AS(1) src / AS(3) dst.
typedef int async_v4i __attribute__((__vector_size__(16)));
typedef __attribute__((address_space(1))) async_v4i* async_gptr;
typedef __attribute__((address_space(3))) async_v4i* async_lptr;
#endif

__device__ inline v8f wmma_f16(v16h a, v16h b, v8f c) {
  // 8 args: (neg_a, A, neg_b, B, c_mod, C, reuse_a, reuse_b)
  return __builtin_amdgcn_wmma_f32_16x16x32_f16(false, a, false, b,
                                                (short)0, c, false, false);
}

// Load a 16x32 operand fragment from a row-major f16 matrix (works for the
// A-operand directly, and for the B-operand when the "rows" of the source are
// the output columns, i.e. weight matrices stored [N, K]).
// Per CDNA5 ISA 16-bit layout: lane m = lane&15, hi = lane>>4;
// element j<8 -> k0 + hi*8 + j ; j>=8 -> k0 + 16 + hi*8 + (j-8).
__device__ inline v16h load_frag(const _Float16* __restrict__ p,
                                 int row0, int k0, int ld, int lane) {
  const int m  = lane & 15;
  const int hi = lane >> 4;
  const _Float16* r = p + (size_t)(row0 + m) * (size_t)ld + k0 + hi * 8;
  v8h lo = *(const v8h*)(r);
  v8h hv = *(const v8h*)(r + 16);
  v16h out;
#pragma unroll
  for (int j = 0; j < 8; ++j) { out[j] = lo[j]; out[j + 8] = hv[j]; }
  return out;
}

// ---------------------------------------------------------------------------
// f32 -> f16 conversion (bandwidth-bound elementwise)
// ---------------------------------------------------------------------------
__global__ void f32_to_f16_kernel(const float* __restrict__ in,
                                  _Float16* __restrict__ out, size_t n) {
  size_t i = (size_t)blockIdx.x * blockDim.x + threadIdx.x;
  size_t stride = (size_t)gridDim.x * blockDim.x;
  for (; i < n; i += stride) out[i] = (_Float16)in[i];
}

// ---------------------------------------------------------------------------
// Generic WMMA GEMM: out[M,Nout] = A[M,K] (row-major f16) x W[Nout,K]^T
// Block = 128 threads (4 waves); each wave does a 16x64 output strip.
// F16OUT: write f16 (projection intermediates). Else: f32 + bias (final).
// ---------------------------------------------------------------------------
template <bool F16OUT>
__global__ __launch_bounds__(128) void gemm_wmma_kernel(
    const _Float16* __restrict__ A, const _Float16* __restrict__ W,
    void* __restrict__ outp, const float* __restrict__ bias,
    int M, int Nout, int K) {
  const int lane = threadIdx.x & 31;
  const int wave = threadIdx.x >> 5;
  const int m    = lane & 15;
  const int hi   = lane >> 4;
  const int row0 = blockIdx.y * 64 + wave * 16;
  const int col0 = blockIdx.x * 64;

  v8f acc[4] = {{}, {}, {}, {}};

  for (int k0 = 0; k0 < K; k0 += 32) {
    v16h af = load_frag(A, row0, k0, K, lane);
    if (k0 + 32 < K)  // hint next A chunk into cache (global_prefetch_b8)
      __builtin_prefetch(A + (size_t)(row0 + m) * K + k0 + 32, 0, 1);
#pragma unroll
    for (int t = 0; t < 4; ++t) {
      v16h bf = load_frag(W, col0 + t * 16, k0, K, lane);
      acc[t] = wmma_f16(af, bf, acc[t]);
    }
  }

#pragma unroll
  for (int t = 0; t < 4; ++t) {
    const int coln = col0 + t * 16 + m;
#pragma unroll
    for (int r = 0; r < 8; ++r) {
      const int row = row0 + r + 8 * hi;
      if constexpr (F16OUT) {
        ((_Float16*)outp)[(size_t)row * Nout + coln] = (_Float16)acc[t][r];
      } else {
        ((float*)outp)[(size_t)row * Nout + coln] = acc[t][r] + bias[coln];
      }
    }
  }
}

// ---------------------------------------------------------------------------
// In-place L2 normalization of 128-wide segments (q rows, k rows).
// One wave per segment; wave32 butterfly reduction.
// ---------------------------------------------------------------------------
__global__ __launch_bounds__(256) void l2norm_kernel(
    _Float16* __restrict__ buf, int nSeg, int segsPerRow, int rowStride) {
  const int wave = threadIdx.x >> 5;
  const int lane = threadIdx.x & 31;
  const int seg  = blockIdx.x * 8 + wave;
  if (seg >= nSeg) return;
  const int row = seg / segsPerRow;
  const int h   = seg % segsPerRow;
  _Float16* p = buf + (size_t)row * rowStride + h * DD;

  float v[4];
#pragma unroll
  for (int j = 0; j < 4; ++j) v[j] = (float)p[lane * 4 + j];
  float ss = v[0] * v[0] + v[1] * v[1] + v[2] * v[2] + v[3] * v[3];
#pragma unroll
  for (int off = 1; off < 32; off <<= 1) ss += __shfl_xor(ss, off, 32);
  const float inv = 1.0f / fmaxf(sqrtf(ss), 1e-12f);
#pragma unroll
  for (int j = 0; j < 4; ++j) p[lane * 4 + j] = (_Float16)(v[j] * inv);
}

// ---------------------------------------------------------------------------
// Flash attention: per (b,h), 64 query rows per workgroup (4 waves x 16 rows).
// QK^T and PV via WMMA; online softmax in the WMMA C/D register layout.
// K-block staging uses GLOBAL_LOAD_ASYNC_TO_LDS_B128 when available.
// ---------------------------------------------------------------------------
__global__ __launch_bounds__(128) void attn_kernel(
    const _Float16* __restrict__ q16, const _Float16* __restrict__ kv16,
    const float* __restrict__ temperature, _Float16* __restrict__ ctx16) {
  // Padded LDS: row strides are multiples of 8 halves (16B -> legal ds b128)
  __shared__ _Float16 Kb[64][136];     // K block, [key][d]
  __shared__ _Float16 Vt[128][72];     // V block transposed, [d][key]
  __shared__ _Float16 Pb[4][16][72];   // per-wave P tile, [qrow][key]

  const int tid  = threadIdx.x;
  const int wave = tid >> 5;
  const int lane = tid & 31;
  const int m    = lane & 15;
  const int hi   = lane >> 4;

  const int bh = blockIdx.y;
  const int b  = bh / HH;
  const int h  = bh % HH;
  const int qrow0 = blockIdx.x * 64 + wave * 16;
  const float temp = temperature[h];

  const _Float16* qbase = q16  + (size_t)b * NN * CC + h * DD;        // ld = CC
  const _Float16* kbase = kv16 + (size_t)b * NN * (2 * CC) + h * DD;  // ld = 2CC
  const _Float16* vbase = kbase + CC;

  // Q operand fragments for this wave's 16 rows (4 k-steps over D=128)
  v16h qf[4];
#pragma unroll
  for (int ks = 0; ks < 4; ++ks)
    qf[ks] = load_frag(qbase, qrow0, ks * 32, CC, lane);

  v8f O[8] = {{}, {}, {}, {}, {}, {}, {}, {}};
  float mrow[8], lrow[8];
#pragma unroll
  for (int r = 0; r < 8; ++r) { mrow[r] = -INFINITY; lrow[r] = 0.0f; }

  for (int kb = 0; kb < NN / 64; ++kb) {
    __syncthreads();  // previous block's LDS reads complete before restage
    // Stage K (row-major, async DMA into LDS) and V (register transpose).
    for (int c = tid; c < 1024; c += 128) {
      const int kr = c >> 4;          // key within block
      const int kc = (c & 15) * 8;    // dim chunk
      const size_t grow = (size_t)(kb * 64 + kr) * (2 * CC);
#if HAVE_ASYNC_LDS
      __builtin_amdgcn_global_load_async_to_lds_b128(
          (async_gptr)(kbase + grow + kc),
          (async_lptr)(&Kb[kr][kc]), 0, 0);
#else
      *(v8h*)(&Kb[kr][kc]) = *(const v8h*)(kbase + grow + kc);
#endif
      v8h vv = *(const v8h*)(vbase + grow + kc);
#pragma unroll
      for (int j = 0; j < 8; ++j) Vt[kc + j][kr] = vv[j];
    }
#if HAVE_ASYNC_LDS
    __builtin_amdgcn_s_wait_asynccnt(0);  // this wave's async copies landed
#endif
    __syncthreads();

    // S = (Q K^T) * temperature : 16 x 64 per wave (4 WMMA tiles)
    v8f S[4];
#pragma unroll
    for (int t = 0; t < 4; ++t) {
      v8f acc = {};
#pragma unroll
      for (int ks = 0; ks < 4; ++ks) {
        v16h bf = load_frag(&Kb[0][0], t * 16, ks * 32, 136, lane);
        acc = wmma_f16(qf[ks], bf, acc);
      }
#pragma unroll
      for (int r = 0; r < 8; ++r) acc[r] *= temp;
      S[t] = acc;
    }

    // Online softmax. C/D layout: VGPR r holds row (r + 8*hi), col = lane&15.
    // Row reduction = xor-butterfly over masks 1,2,4,8 within each 16-lane half.
#pragma unroll
    for (int r = 0; r < 8; ++r) {
      float rmax = S[0][r];
#pragma unroll
      for (int t = 1; t < 4; ++t) rmax = fmaxf(rmax, S[t][r]);
#pragma unroll
      for (int off = 1; off < 16; off <<= 1)
        rmax = fmaxf(rmax, __shfl_xor(rmax, off, 32));

      const float nm = fmaxf(mrow[r], rmax);
      const float sc = __expf(mrow[r] - nm);  // exp(-inf) = 0 on first block
      mrow[r] = nm;

      float rsum = 0.0f;
#pragma unroll
      for (int t = 0; t < 4; ++t) {
        const float pv = __expf(S[t][r] - nm);
        S[t][r] = pv;
        rsum += pv;
      }
#pragma unroll
      for (int off = 1; off < 16; off <<= 1) rsum += __shfl_xor(rsum, off, 32);
      lrow[r] = lrow[r] * sc + rsum;
#pragma unroll
      for (int td = 0; td < 8; ++td) O[td][r] *= sc;
    }

    // P -> per-wave LDS, then reload in the A-operand layout for PV.
#pragma unroll
    for (int t = 0; t < 4; ++t)
#pragma unroll
      for (int r = 0; r < 8; ++r)
        Pb[wave][r + 8 * hi][t * 16 + m] = (_Float16)S[t][r];

    // O += P(16x64) x V(64x128): 2 k-steps, 8 d-tiles.
#pragma unroll
    for (int ks2 = 0; ks2 < 2; ++ks2) {
      v16h pa = load_frag(&Pb[wave][0][0], 0, ks2 * 32, 72, lane);
#pragma unroll
      for (int td = 0; td < 8; ++td) {
        v16h vb = load_frag(&Vt[0][0], td * 16, ks2 * 32, 72, lane);
        O[td] = wmma_f16(pa, vb, O[td]);
      }
    }
  }

  // Normalize by the softmax denominator and store ctx in [B,N,H*D] f16.
  float inv[8];
#pragma unroll
  for (int r = 0; r < 8; ++r) inv[r] = 1.0f / lrow[r];
  _Float16* obase = ctx16 + ((size_t)b * NN + qrow0) * CC + h * DD;
#pragma unroll
  for (int td = 0; td < 8; ++td)
#pragma unroll
    for (int r = 0; r < 8; ++r)
      obase[(size_t)(r + 8 * hi) * CC + td * 16 + m] =
          (_Float16)(O[td][r] * inv[r]);
}

// ---------------------------------------------------------------------------
// Orchestration
// ---------------------------------------------------------------------------
extern "C" void kernel_launch(void* const* d_in, const int* in_sizes, int n_in,
                              void* d_out, int out_size, void* d_ws,
                              size_t ws_size, hipStream_t stream) {
  const float* x     = (const float*)d_in[0];
  const float* y     = (const float*)d_in[1];
  const float* Wq    = (const float*)d_in[2];
  const float* Wkv   = (const float*)d_in[3];
  const float* Wproj = (const float*)d_in[4];
  const float* bproj = (const float*)d_in[5];
  const float* temp  = (const float*)d_in[6];

  const size_t NTOK = (size_t)BB * NN;        // 8192 rows
  const size_t NX   = NTOK * CC;              // 6291456 elements

  char* ws = (char*)d_ws;
  size_t off = 0;
  auto alloc = [&](size_t bytes) -> char* {
    char* p = ws + off;
    off += (bytes + 255) & ~(size_t)255;
    return p;
  };
  _Float16* xh   = (_Float16*)alloc(NX * 2);
  _Float16* yh   = (_Float16*)alloc(NX * 2);
  _Float16* wqh  = (_Float16*)alloc((size_t)CC * CC * 2);
  _Float16* wkvh = (_Float16*)alloc((size_t)2 * CC * CC * 2);
  _Float16* wph  = (_Float16*)alloc((size_t)CC * CC * 2);
  _Float16* q16  = (_Float16*)alloc(NX * 2);
  _Float16* kv16 = (_Float16*)alloc(NX * 2 * 2);
  _Float16* c16  = (_Float16*)alloc(NX * 2);

  // 1) precision conversion
  f32_to_f16_kernel<<<1024, 256, 0, stream>>>(x, xh, NX);
  f32_to_f16_kernel<<<1024, 256, 0, stream>>>(y, yh, NX);
  f32_to_f16_kernel<<<256, 256, 0, stream>>>(Wq, wqh, (size_t)CC * CC);
  f32_to_f16_kernel<<<512, 256, 0, stream>>>(Wkv, wkvh, (size_t)2 * CC * CC);
  f32_to_f16_kernel<<<256, 256, 0, stream>>>(Wproj, wph, (size_t)CC * CC);

  // 2) projections (WMMA)
  gemm_wmma_kernel<true><<<dim3(CC / 64, NTOK / 64), 128, 0, stream>>>(
      xh, wqh, q16, nullptr, (int)NTOK, CC, CC);
  gemm_wmma_kernel<true><<<dim3(2 * CC / 64, NTOK / 64), 128, 0, stream>>>(
      yh, wkvh, kv16, nullptr, (int)NTOK, 2 * CC, CC);

  // 3) L2 normalize q and k (in place, one wave per 128-wide segment)
  const int nSeg = (int)NTOK * HH;  // 49152
  l2norm_kernel<<<(nSeg + 7) / 8, 256, 0, stream>>>(q16, nSeg, HH, CC);
  l2norm_kernel<<<(nSeg + 7) / 8, 256, 0, stream>>>(kv16, nSeg, HH, 2 * CC);

  // 4) flash attention (WMMA QK^T + PV, online softmax)
  attn_kernel<<<dim3(NN / 64, BB * HH), 128, 0, stream>>>(q16, kv16, temp, c16);

  // 5) output projection + bias -> f32 d_out
  gemm_wmma_kernel<false><<<dim3(CC / 64, NTOK / 64), 128, 0, stream>>>(
      c16, wph, d_out, bproj, (int)NTOK, CC, CC);
}
